// MoE_17214228922764
// MI455X (gfx1250) — matrix operand order repaired and checked by reading, not compiled, not run on hardware
//
#include <hip/hip_runtime.h>
#include <hip/hip_bf16.h>
#include <math.h>

// ---------------- problem constants (match reference) ----------------
#define T_TOK  (4 * 2048)   // B*S tokens
#define DDIM   1280
#define IDIM   1280
#define NEXP   63
#define TOPK   7
#define TM     32           // tokens per block tile

typedef __attribute__((ext_vector_type(16))) __bf16 v16bf;
typedef __attribute__((ext_vector_type(8)))  __bf16 v8bf;
typedef __attribute__((ext_vector_type(8)))  float  v8f;

typedef unsigned int uint32x4 __attribute__((ext_vector_type(4)));
typedef int          int32x4  __attribute__((ext_vector_type(4)));
typedef int          int32x8  __attribute__((ext_vector_type(8)));

// ---------------------------------------------------------------------
// Kernel 1a: fp32 -> bf16 conversion of x (A-operand staging for WMMA)
// ---------------------------------------------------------------------
__global__ void cvt_kernel(const float* __restrict__ x, __bf16* __restrict__ xbf, int n) {
    int i = blockIdx.x * blockDim.x + threadIdx.x;
    if (i < n) xbf[i] = (__bf16)x[i];
}

// Kernel 1b: fp32 [nexp][K][N] -> bf16 transposed [nexp][N][K]
// (puts the WMMA B-fragment's 16 consecutive K values contiguously in memory)
__global__ void cvt_transpose_kernel(const float* __restrict__ W, __bf16* __restrict__ WT,
                                     int K, int N, long long total) {
    long long i = (long long)blockIdx.x * blockDim.x + threadIdx.x;
    if (i >= total) return;
    int n = (int)(i % N);
    long long t = i / N;
    int k = (int)(t % K);
    long long e = t / K;
    WT[e * (long long)K * N + (long long)n * K + k] = (__bf16)W[i];
}

__global__ void zero_counts_kernel(int* __restrict__ counts) {
    if (threadIdx.x < 64) counts[threadIdx.x] = 0;
}

// ---------------------------------------------------------------------
// Kernel 2: router — softmax over 63 experts, top-7, scatter into
// per-expert (token, gate) lists. One 64-thread block per token.
// ---------------------------------------------------------------------
__global__ __launch_bounds__(64) void router_kernel(
    const float* __restrict__ x, const float* __restrict__ wr,
    const float* __restrict__ br, int* __restrict__ counts,
    int* __restrict__ tokList, float* __restrict__ gateList) {
    __shared__ float sx[DDIM];
    __shared__ float sl[NEXP];

    const int tok = blockIdx.x;
    const float* xr = x + (size_t)tok * DDIM;
    for (int i = threadIdx.x; i < DDIM; i += blockDim.x) sx[i] = xr[i];
    __syncthreads();

    const int e = threadIdx.x;
    if (e < NEXP) {
        float acc = br[e];
        for (int d = 0; d < DDIM; ++d)
            acc = fmaf(sx[d], wr[(size_t)d * NEXP + e], acc);
        sl[e] = acc;
    }
    __syncthreads();

    if (threadIdx.x == 0) {
        float mx = sl[0];
        for (int i = 1; i < NEXP; ++i) mx = fmaxf(mx, sl[i]);
        float sum = 0.f;
        for (int i = 0; i < NEXP; ++i) { float p = __expf(sl[i] - mx); sl[i] = p; sum += p; }
        const float inv = 1.0f / sum;
        for (int k = 0; k < TOPK; ++k) {
            int bi = 0; float bv = -1.f;
            for (int i = 0; i < NEXP; ++i) if (sl[i] > bv) { bv = sl[i]; bi = i; }
            int pos = atomicAdd(&counts[bi], 1);
            tokList[(size_t)bi * T_TOK + pos]  = tok;
            gateList[(size_t)bi * T_TOK + pos] = bv * inv;
            sl[bi] = -2.f;
        }
    }
}

// ---------------------------------------------------------------------
// Kernel 3/4: fused expert FFN  out += gate * (GELU(x@W1+b1)@W2 + b2)
// 256 threads = 8 waves in a 2x4 grid of 16x16 WMMA tiles over a
// 32-token x 64-column macro tile. A gathered via TDM gather-mode
// tensor_load_to_lds; hidden tile (bf16, GELU applied) staged in LDS.
// BF16W=true : weights pre-converted to transposed bf16 (contiguous B frags)
// BF16W=false: strided fp32 weight loads + on-the-fly cvt (fallback)
// ---------------------------------------------------------------------
template <bool BF16W>
__global__ __launch_bounds__(256) void moe_ffn_kernel(
    const __bf16* __restrict__ xbf,
    const float*  __restrict__ x,         // residual source (shared mode)
    const void*   __restrict__ W1,        // BF16W ? [(E,)I,D] bf16 : [(E,)D,I] f32
    const float*  __restrict__ B1,        // [(E,)I]
    const void*   __restrict__ W2,        // BF16W ? [(E,)D,I] bf16 : [(E,)I,D] f32
    const float*  __restrict__ B2,        // [(E,)D]
    const int*    __restrict__ counts,    // routed only
    const int*    __restrict__ tokList,
    const float*  __restrict__ gateList,
    float*        __restrict__ out,
    int is_shared) {
    extern __shared__ __align__(16) char smem[];
    __bf16* ldsX  = (__bf16*)smem;                  // [TM][DDIM] (TDM packs rows densely)
    __bf16* ldsH  = ldsX + (size_t)TM * DDIM;       // [TM][IDIM]
    int*    sTok  = (int*)(ldsH + (size_t)TM * IDIM);
    float*  sGate = (float*)(sTok + TM);

    const int e    = blockIdx.x;
    const int cnt  = is_shared ? T_TOK : counts[e];
    const int row0 = blockIdx.y * TM;
    if (row0 >= cnt) return;                        // uniform per block: EXEC stays full

    const size_t eoW = is_shared ? 0 : (size_t)e * DDIM * IDIM;   // same elem count both layouts
    const float* b1 = B1 + (is_shared ? 0 : (size_t)e * IDIM);
    const float* b2 = B2 + (is_shared ? 0 : (size_t)e * DDIM);

    const int tid  = threadIdx.x;
    const int lane = tid & 31;
    const int wave = tid >> 5;
    const int wm   = wave >> 2;             // 0..1 : 16-row tile
    const int wn   = wave & 3;              // 0..3 : 16-col tile within 64-col chunk
    const int lrow = lane & 15;
    const int kA   = (lane < 16) ? 0 : 16;  // A fragment K-half (contiguous, matches B)
    const int kB   = (lane < 16) ? 0 : 16;  // B fragment K-half (contiguous, matches A)

    // uniform token index for a tile row (scalar loads: address is wave-uniform)
    auto tokOf = [&](int i) -> int {
        const int r  = row0 + i;
        const int rr = (r < cnt) ? r : (cnt - 1);
        return is_shared ? rr : tokList[(size_t)e * T_TOK + rr];
    };

    // ---- (token, gate) sideband for the epilogue ----
    if (tid < TM) {
        const int r  = row0 + tid;
        const int rr = (r < cnt) ? r : (cnt - 1);
        sTok[tid]  = tokOf(tid);
        sGate[tid] = is_shared ? 1.0f
                               : ((r < cnt) ? gateList[(size_t)e * T_TOK + rr] : 0.0f);
    }

    // ---- TDM gather: 2 x (16 rows x 1280 bf16) from xbf -> ldsX ----
    if (wave == 0) {
        const unsigned long long gaddr = (unsigned long long)(const void*)xbf;
        const unsigned ldsBase = (unsigned)__builtin_amdgcn_groupstaticsize();
        #pragma unroll
        for (int half = 0; half < 2; ++half) {
            uint32x4 g0;
            g0[0] = 1u | (1u << 31);                           // count=1, gather_mode=1, idx16
            g0[1] = ldsBase + (unsigned)(half * 16 * DDIM * 2);// lds_addr (bytes)
            g0[2] = (unsigned)(gaddr & 0xFFFFFFFFu);           // global_addr[31:0]
            g0[3] = (unsigned)((gaddr >> 32) & 0x1FFFFFFu) | (2u << 30);  // addr[56:32] | type=2
            int32x8 g1;
            g1[0] = (1 << 16);                                 // data_size = 2B, no flags
            g1[1] = (int)((DDIM & 0xFFFF) << 16);              // tensor_dim0 lo16
            g1[2] = (int)((DDIM >> 16) | ((T_TOK & 0xFFFF) << 16)); // dim0 hi | dim1 lo
            g1[3] = (int)((T_TOK >> 16) | (DDIM << 16));       // dim1 hi | tile_dim0 = 1280
            g1[4] = 16;                                        // tile_dim1 = #valid indices
            g1[5] = DDIM;                                      // tensor_dim0_stride lo32
            g1[6] = 0;
            g1[7] = 0;
            int32x4 g2, g3;
            #pragma unroll
            for (int j = 0; j < 4; ++j) {
                const int a0 = half * 16 + j * 2;
                const int b0 = half * 16 + 8 + j * 2;
                g2[j] = (tokOf(a0) & 0xFFFF) | ((tokOf(a0 + 1) & 0xFFFF) << 16);
                g3[j] = (tokOf(b0) & 0xFFFF) | ((tokOf(b0 + 1) & 0xFFFF) << 16);
            }
#if defined(__clang_major__) && (__clang_major__ >= 23)
            int32x8 gz = {0, 0, 0, 0, 0, 0, 0, 0};
            __builtin_amdgcn_tensor_load_to_lds(g0, g1, g2, g3, gz, 0);
#else
            __builtin_amdgcn_tensor_load_to_lds(g0, g1, g2, g3, 0);
#endif
        }
        __builtin_amdgcn_s_wait_tensorcnt(0);
    }
    __syncthreads();

    // ---- stage 1: hidden = GELU(x @ W1 + b1), bf16 into LDS ----
    for (int nc = 0; nc < IDIM / 64; ++nc) {
        const int nAbs = nc * 64 + wn * 16 + lrow;
        v8f c = {};
        for (int kb = 0; kb < DDIM; kb += 32) {
            const __bf16* ar = ldsX + (size_t)(wm * 16 + lrow) * DDIM + kb + kA;
            const v8bf lo = *(const v8bf*)ar;
            const v8bf hi = *(const v8bf*)(ar + 8);
            v16bf a;
            #pragma unroll
            for (int j = 0; j < 8; ++j) { a[j] = lo[j]; a[j + 8] = hi[j]; }

            v16bf b;
            if (BF16W) {
                const __bf16* wp = (const __bf16*)W1 + eoW + (size_t)nAbs * DDIM + kb + kB;
                if (kb + 32 < DDIM) __builtin_prefetch(wp + 48, 0, 0);
                b = *(const v16bf*)wp;                  // 16 consecutive K: 2x b128
            } else {
                const float* wp = (const float*)W1 + eoW + (size_t)(kb + kB) * IDIM + nAbs;
                if (kb + 32 < DDIM) __builtin_prefetch(wp + 32 * (size_t)IDIM, 0, 0);
                #pragma unroll
                for (int j = 0; j < 16; ++j) b[j] = (__bf16)wp[(size_t)j * IDIM];
            }
            c = __builtin_amdgcn_wmma_f32_16x16x32_bf16(
                    false, a, false, b, (short)0, c, false, false);
        }
        const float bias = b1[nAbs];
        #pragma unroll
        for (int r = 0; r < 8; ++r) {
            const int m = wm * 16 + r + ((lane < 16) ? 0 : 8);
            const float v = c[r] + bias;
            const float g = 0.5f * v * (1.0f + erff(v * 0.70710678118654752f));
            ldsH[(size_t)m * IDIM + nAbs] = (__bf16)g;
        }
    }
    __syncthreads();

    // ---- stage 2: out_tile = hidden @ W2 + b2 ----
    for (int nc = 0; nc < DDIM / 64; ++nc) {
        const int nAbs = nc * 64 + wn * 16 + lrow;
        v8f c = {};
        for (int kb = 0; kb < IDIM; kb += 32) {
            const __bf16* ar = ldsH + (size_t)(wm * 16 + lrow) * IDIM + kb + kA;
            const v8bf lo = *(const v8bf*)ar;
            const v8bf hi = *(const v8bf*)(ar + 8);
            v16bf a;
            #pragma unroll
            for (int j = 0; j < 8; ++j) { a[j] = lo[j]; a[j + 8] = hi[j]; }

            v16bf b;
            if (BF16W) {
                const __bf16* wp = (const __bf16*)W2 + eoW + (size_t)nAbs * IDIM + kb + kB;
                if (kb + 32 < IDIM) __builtin_prefetch(wp + 48, 0, 0);
                b = *(const v16bf*)wp;
            } else {
                const float* wp = (const float*)W2 + eoW + (size_t)(kb + kB) * DDIM + nAbs;
                if (kb + 32 < IDIM) __builtin_prefetch(wp + 32 * (size_t)DDIM, 0, 0);
                #pragma unroll
                for (int j = 0; j < 16; ++j) b[j] = (__bf16)wp[(size_t)j * DDIM];
            }
            c = __builtin_amdgcn_wmma_f32_16x16x32_bf16(
                    false, a, false, b, (short)0, c, false, false);
        }
        const float bias = b2[nAbs];
        #pragma unroll
        for (int r = 0; r < 8; ++r) {
            const int m   = wm * 16 + r + ((lane < 16) ? 0 : 8);
            const int tok = sTok[m];
            const float g = sGate[m];
            const float val = c[r] + bias;
            const size_t oidx = (size_t)tok * DDIM + nAbs;
            if (is_shared) out[oidx] = x[oidx] + val;          // residual + shared, plain store
            else if (g != 0.f) atomicAdd(&out[oidx], g * val); // routed accumulation
        }
    }
}

// ---------------------------------------------------------------------
extern "C" void kernel_launch(void* const* d_in, const int* in_sizes, int n_in,
                              void* d_out, int out_size, void* d_ws, size_t ws_size,
                              hipStream_t stream) {
    const float* x   = (const float*)d_in[0];
    const float* sw1 = (const float*)d_in[1];
    const float* sb1 = (const float*)d_in[2];
    const float* sw2 = (const float*)d_in[3];
    const float* sb2 = (const float*)d_in[4];
    const float* rw1 = (const float*)d_in[5];
    const float* rb1 = (const float*)d_in[6];
    const float* rw2 = (const float*)d_in[7];
    const float* rb2 = (const float*)d_in[8];
    const float* wr  = (const float*)d_in[9];
    const float* br  = (const float*)d_in[10];
    float* out = (float*)d_out;

    // -------- workspace layout --------
    const size_t SZ_XBF  = (size_t)T_TOK * DDIM * 2;
    const size_t SZ_LIST = (size_t)NEXP * T_TOK * 4;
    const size_t SZ_WEXP = (size_t)DDIM * IDIM * 2;        // one bf16 matrix per expert
    char* ws = (char*)d_ws;
    int*    counts   = (int*)ws;                                       // 1 KB slot
    __bf16* xbf      = (__bf16*)(void*)(ws + 1024);
    int*    tokList  = (int*)(ws + 1024 + SZ_XBF);
    float*  gateList = (float*)(ws + 1024 + SZ_XBF + SZ_LIST);
    const size_t off_big = 1024 + SZ_XBF + 2 * SZ_LIST;
    __bf16* swT1 = (__bf16*)(void*)(ws + off_big);
    __bf16* swT2 = (__bf16*)(void*)(ws + off_big + SZ_WEXP);
    __bf16* rwT1 = (__bf16*)(void*)(ws + off_big + 2 * SZ_WEXP);
    __bf16* rwT2 = (__bf16*)(void*)(ws + off_big + (2 + (size_t)NEXP) * SZ_WEXP);
    const size_t req_big = off_big + (2 + 2 * (size_t)NEXP) * SZ_WEXP; // ~445 MB
    const bool big = (ws_size >= req_big);

    const int nX = T_TOK * DDIM;
    cvt_kernel<<<(nX + 255) / 256, 256, 0, stream>>>(x, xbf, nX);
    zero_counts_kernel<<<1, 64, 0, stream>>>(counts);

    if (big) {  // pre-convert weights once: fp32 row-major -> bf16 transposed
        const long long tS = (long long)DDIM * IDIM;
        const long long tR = (long long)NEXP * DDIM * IDIM;
        const int blk = 256;
        cvt_transpose_kernel<<<(unsigned)((tS + blk - 1) / blk), blk, 0, stream>>>(sw1, swT1, DDIM, IDIM, tS);
        cvt_transpose_kernel<<<(unsigned)((tS + blk - 1) / blk), blk, 0, stream>>>(sw2, swT2, IDIM, DDIM, tS);
        cvt_transpose_kernel<<<(unsigned)((tR + blk - 1) / blk), blk, 0, stream>>>(rw1, rwT1, DDIM, IDIM, tR);
        cvt_transpose_kernel<<<(unsigned)((tR + blk - 1) / blk), blk, 0, stream>>>(rw2, rwT2, IDIM, DDIM, tR);
    }

    router_kernel<<<T_TOK, 64, 0, stream>>>(x, wr, br, counts, tokList, gateList);

    const size_t shmem = (size_t)TM * (DDIM + IDIM) * sizeof(__bf16)
                       + TM * (sizeof(int) + sizeof(float));
    dim3 gs(1, T_TOK / TM);
    dim3 gr(NEXP, T_TOK / TM);

    if (big) {
        moe_ffn_kernel<true><<<gs, 256, shmem, stream>>>(xbf, x, swT1, sb1, swT2, sb2,
                                                         nullptr, nullptr, nullptr, out, 1);
        moe_ffn_kernel<true><<<gr, 256, shmem, stream>>>(xbf, x, rwT1, rb1, rwT2, rb2,
                                                         counts, tokList, gateList, out, 0);
    } else {
        moe_ffn_kernel<false><<<gs, 256, shmem, stream>>>(xbf, x, sw1, sb1, sw2, sb2,
                                                          nullptr, nullptr, nullptr, out, 1);
        moe_ffn_kernel<false><<<gr, 256, shmem, stream>>>(xbf, x, rw1, rb1, rw2, rb2,
                                                          counts, tokList, gateList, out, 0);
    }
}
